// NSC_59133109732095
// MI455X (gfx1250) — matrix operand order repaired and checked by reading, not compiled, not run on hardware
//
#include <hip/hip_runtime.h>

// ---------------------------------------------------------------------------
// Fused NSC kernel for MI455X (gfx1250, wave32, WMMA).
//   per half-complex (128 grid points): features -> L1(18->128,sin)
//   -> L2(128->128,sin) -> L3(128->3). Activations AND f16-converted,
//   zero-padded weights live in LDS; GEMMs use v_wmma_f32_16x16x32_f16.
//   All WMMA operand fetches are aligned ds_load_b128 pairs.
// ---------------------------------------------------------------------------

typedef __attribute__((ext_vector_type(16))) _Float16 v16h;
typedef __attribute__((ext_vector_type(8)))  _Float16 v8h;
typedef __attribute__((ext_vector_type(8)))  float    v8f;

#define HID   128
#define DIN   18      // 15 encodings + 3 sin(p)
#define FSTR  40      // 32-wide rows padded to 80B (bank-conflict dodge)
#define ASTR  136     // 128-wide rows padded to 272B

union HFrag { v16h v; v8h h[2]; };

// A-matrix fragment (16x32 f16), contiguous-k order (B uses the same):
//   lane l: row m = l%16; 16 contiguous K halves at k0 = (l/16)*16
__device__ __forceinline__ v16h load_afrag(const _Float16* rowp, int k0) {
  HFrag a;
  a.h[0] = *(const v8h*)(rowp + k0);
  a.h[1] = *(const v8h*)(rowp + k0 + 8);
  return a.v;
}

// B-matrix fragment (32x16 f16) from an f16 LDS weight row Wt[n][*]:
//   lane l: col n = l%16; 16 contiguous K halves at k1 = (l/16)*16
__device__ __forceinline__ v16h load_bfrag(const _Float16* rowp, int k1) {
  HFrag b;
  b.h[0] = *(const v8h*)(rowp + k1);
  b.h[1] = *(const v8h*)(rowp + k1 + 8);
  return b.v;
}

__device__ __forceinline__ v8f vzero8() {
  v8f z;
#pragma unroll
  for (int i = 0; i < 8; ++i) z[i] = 0.0f;
  return z;
}

__global__ __launch_bounds__(256)
void nsc_fused_mlp(const float* __restrict__ points,
                   const float* __restrict__ enc,
                   const float* __restrict__ W1, const float* __restrict__ b1,
                   const float* __restrict__ W2, const float* __restrict__ b2,
                   const float* __restrict__ W3, const float* __restrict__ b3,
                   const int*   __restrict__ cplx,
                   float* __restrict__ out, int C)
{
  __shared__ __align__(16) _Float16 s_feat[128 * FSTR]; // layer-1 A (K padded to 32)
  __shared__ __align__(16) _Float16 s_act [128 * ASTR]; // act1/act2 in place (wave-private rows)
  __shared__ __align__(16) _Float16 s_w1  [128 * FSTR]; // W1 f16, K 18->32 zero-padded
  __shared__ __align__(16) _Float16 s_w2  [128 * ASTR]; // W2 f16
  __shared__ __align__(16) _Float16 s_w3  [ 16 * ASTR]; // W3 f16, N 3->16 zero-padded
  __shared__ __align__(16) float    s_p   [128 * 4];    // interpolated coords for p + X

  const int tid  = threadIdx.x;
  const int lane = tid & 31;
  const int wave = tid >> 5;                 // 8 waves; wave owns rows [16w, 16w+16)
  const int c    = blockIdx.x >> 1;          // complex index
  const int g0   = (blockIdx.x & 1) * 128;   // which half of the 256-point grid

  // ---- phase A: zero the padded weight regions (uniform stores, no branches)
  for (int idx = tid; idx < 128 * 32; idx += 256) {
    const int n = idx >> 5, k = idx & 31;
    s_w1[n * FSTR + k] = (_Float16)0.0f;
  }
  for (int idx = tid; idx < 16 * 128; idx += 256) {
    const int n = idx >> 7, k = idx & 127;
    s_w3[n * ASTR + k] = (_Float16)0.0f;
  }
  __syncthreads();

  // ---- phase B: stage f16 weights + compute features ----------------------
  for (int idx = tid; idx < 128 * DIN; idx += 256) {
    const int n = idx / DIN, k = idx - n * DIN;
    s_w1[n * FSTR + k] = (_Float16)W1[idx];
  }
  for (int idx = tid; idx < 128 * HID; idx += 256) {
    const int n = idx >> 7, k = idx & 127;
    s_w2[n * ASTR + k] = (_Float16)W2[idx];
  }
  for (int idx = tid; idx < 3 * HID; idx += 256) {
    const int n = idx >> 7, k = idx & 127;
    s_w3[n * ASTR + k] = (_Float16)W3[idx];
  }

  if (tid < 128) {  // 1 thread = 1 grid point
    const int g = g0 + tid;
    const int i = g >> 4, j = g & 15;
    const float u = (float)i * (1.0f / 15.0f);
    const float v = (float)j * (1.0f / 15.0f);
    // einsum('ia,jb,cabf'): corner(a,b) = cplx[c][2a+b], weights (1-u,u)x(1-v,v)
    const float w00 = (1.0f - u) * (1.0f - v);
    const float w01 = (1.0f - u) * v;
    const float w10 = u * (1.0f - v);
    const float w11 = u * v;
    const int i0 = cplx[c * 4 + 0], i1 = cplx[c * 4 + 1];
    const int i2 = cplx[c * 4 + 2], i3 = cplx[c * 4 + 3];
    _Float16* frow = s_feat + tid * FSTR;
#pragma unroll
    for (int f = 0; f < 15; ++f) {
      float e = w00 * enc[i0 * 15 + f] + w01 * enc[i1 * 15 + f]
              + w10 * enc[i2 * 15 + f] + w11 * enc[i3 * 15 + f];
      frow[f] = (_Float16)e;
    }
#pragma unroll
    for (int d = 0; d < 3; ++d) {
      float p = w00 * points[i0 * 3 + d] + w01 * points[i1 * 3 + d]
              + w10 * points[i2 * 3 + d] + w11 * points[i3 * 3 + d];
      s_p[tid * 4 + d] = p;
      frow[15 + d] = (_Float16)__sinf(p);
    }
#pragma unroll
    for (int k = DIN; k < 32; ++k) frow[k] = (_Float16)0.0f;  // zero K padding
  }
  __syncthreads();

  // per-lane WMMA addressing
  const int arow = wave * 16 + (lane & 15);   // A-fragment row (local)
  const int k0a  = (lane >> 4) * 16;          // A k-offset within 32-chunk
  const int k0b  = (lane >> 4) * 16;          // B k-offset within 32-chunk
  const int ncol = lane & 15;                 // C/D column; B column
  const int mhi  = (lane >> 4) * 8;           // C/D row offset: m = v + mhi

  const _Float16* featrow = s_feat + arow * FSTR;
  const _Float16* actrow  = s_act  + arow * ASTR;
  const _Float16* w3row   = s_w3   + ncol * ASTR;

  // ---------------- layer 1: act = sin(feat @ W1^T + b1), K=32 (padded) ----
  {
    v16h a = load_afrag(featrow, k0a);
#pragma unroll
    for (int nt = 0; nt < 8; ++nt) {
      const int n = nt * 16 + ncol;
      v16h b = load_bfrag(s_w1 + n * FSTR, k0b);
      v8f acc = vzero8();
      acc = __builtin_amdgcn_wmma_f32_16x16x32_f16(false, a, false, b,
                                                   (short)0, acc, false, false);
      const float bias = b1[n];
#pragma unroll
      for (int r = 0; r < 8; ++r) {
        const int m = wave * 16 + r + mhi;    // D layout: row = vgpr + 8*(lane/16)
        s_act[m * ASTR + n] = (_Float16)__sinf(acc[r] + bias);
      }
    }
  }
  __syncthreads();

  // ---------------- layer 2: act = sin(act @ W2^T + b2), K=128 -------------
  {
    v8f acc[8];
#pragma unroll
    for (int nt = 0; nt < 8; ++nt) acc[nt] = vzero8();
#pragma unroll
    for (int kc = 0; kc < 4; ++kc) {
      v16h a = load_afrag(actrow, kc * 32 + k0a);
#pragma unroll
      for (int nt = 0; nt < 8; ++nt) {
        const int n = nt * 16 + ncol;
        v16h b = load_bfrag(s_w2 + n * ASTR, kc * 32 + k0b);
        acc[nt] = __builtin_amdgcn_wmma_f32_16x16x32_f16(false, a, false, b,
                                                         (short)0, acc[nt], false, false);
      }
    }
    // rows are wave-private: safe to overwrite act1 with act2 in place
#pragma unroll
    for (int nt = 0; nt < 8; ++nt) {
      const int n = nt * 16 + ncol;
      const float bias = b2[n];
#pragma unroll
      for (int r = 0; r < 8; ++r) {
        const int m = wave * 16 + r + mhi;
        s_act[m * ASTR + n] = (_Float16)__sinf(acc[nt][r] + bias);
      }
    }
  }
  __syncthreads();

  // ---------------- layer 3: X = act @ W3^T + b3 (N padded 3 -> 16) --------
  {
    v8f acc = vzero8();
#pragma unroll
    for (int kc = 0; kc < 4; ++kc) {
      v16h a = load_afrag(actrow, kc * 32 + k0a);
      v16h b = load_bfrag(w3row, kc * 32 + k0b);  // rows 3..15 are zeros
      acc = __builtin_amdgcn_wmma_f32_16x16x32_f16(false, a, false, b,
                                                   (short)0, acc, false, false);
    }
    if (ncol < 3) {
      const float bias = b3[ncol];
      const long long base0 = (long long)c * (256 * 3);
      const long long xoff  = (long long)C * 768;  // second tuple element
#pragma unroll
      for (int r = 0; r < 8; ++r) {
        const int mloc = wave * 16 + r + mhi;
        const int g    = g0 + mloc;
        const float x  = acc[r] + bias;
        const float pv = s_p[mloc * 4 + ncol];
        const long long o = base0 + (long long)g * 3 + ncol;
        out[o]        = pv + x;   // p + X
        out[xoff + o] = x;        // X
      }
    }
  }

  if (blockIdx.x == 0 && tid == 0) {
    out[(long long)C * 768 * 2] = 1.0f;  // lipschitz scalar
  }
}

extern "C" void kernel_launch(void* const* d_in, const int* in_sizes, int n_in,
                              void* d_out, int out_size, void* d_ws, size_t ws_size,
                              hipStream_t stream) {
  const float* points = (const float*)d_in[0];
  const float* enc    = (const float*)d_in[1];
  const float* W1     = (const float*)d_in[2];
  const float* b1     = (const float*)d_in[3];
  const float* W2     = (const float*)d_in[4];
  const float* b2     = (const float*)d_in[5];
  const float* W3     = (const float*)d_in[6];
  const float* b3     = (const float*)d_in[7];
  const int*   cplx   = (const int*)d_in[8];
  // d_in[9] = resolution (16), baked into the kernel's tiling
  const int C = in_sizes[8] / 4;
  float* out = (float*)d_out;

  nsc_fused_mlp<<<C * 2, 256, 0, stream>>>(points, enc, W1, b1, W2, b2,
                                           W3, b3, cplx, out, C);
}
